// GRU_31035433681036
// MI455X (gfx1250) — matrix operand order repaired and checked by reading, not compiled, not run on hardware
//
#include <hip/hip_runtime.h>
#include <stdint.h>

// ---------------------------------------------------------------------------
// Problem constants
// ---------------------------------------------------------------------------
#define Bq 512
#define Tq 256
#define Iq 256
#define Hq 1024
#define Oq 64

#define BT   16      // batch tile per workgroup
#define CSTR 1288    // comb LDS row stride (bf16 elems): 16B aligned, bank-spread
#define HSTR 1032    // head LDS row stride
#define KT_G 40      // K tiles (of 32) for 1280-deep recurrent GEMMs
#define KT_H 32      // K tiles for 1024-deep head GEMMs

#define TILE_G (KT_G << 9)   // elems between consecutive n-tiles (recurrent)
#define TILE_H (KT_H << 9)   // elems between consecutive n-tiles (head)

// ws layout (bytes)
#define WZ_OFF 0u
#define WR_OFF 2621440u
#define WH_OFF 5242880u
#define W1_OFF 7864320u
#define W2_OFF 9961472u
#define H_OFF  10092544u
// total ~12.2 MB

typedef __attribute__((ext_vector_type(16))) __bf16 v16bf;
typedef __attribute__((ext_vector_type(8)))  float  v8f;

struct Bytes32 { uint4 lo, hi; };

__device__ __forceinline__ uint16_t f2b(float f) {
  union { float f; uint32_t u; } c; c.f = f;
  uint32_t u = c.u;
  uint32_t r = (u + 0x7FFFu + ((u >> 16) & 1u)) >> 16;  // RNE
  return (uint16_t)r;
}

// B-operand fragment from pre-swizzled global memory: 32 contiguous bytes/lane
__device__ __forceinline__ v16bf load_bfrag(const uint16_t* __restrict__ p) {
  Bytes32 b;
  b.lo = *(const uint4*)(p);
  b.hi = *(const uint4*)(p + 8);
  return __builtin_bit_cast(v16bf, b);
}

// A-operand fragment from LDS row (16-bit A layout: half-waves interleave K
// chunks {0..7,16..23} / {8..15,24..31}); k0 = kt*32 + half*8, both 16B aligned
__device__ __forceinline__ v16bf load_afrag(const uint16_t* row, int k0) {
  Bytes32 b;
  b.lo = *(const uint4*)(row + k0);
  b.hi = *(const uint4*)(row + k0 + 16);
  return __builtin_bit_cast(v16bf, b);
}

__device__ __forceinline__ v8f wmma_bf16(v16bf a, v16bf b, v8f c) {
  return __builtin_amdgcn_wmma_f32_16x16x32_bf16(
      /*neg_a=*/false, a, /*neg_b=*/false, b,
      /*c_mod=*/(short)0, c, /*reuse_a=*/false, /*reuse_b=*/false);
}

__device__ __forceinline__ float sigmoidf_(float x) {
  return 1.0f / (1.0f + __expf(-x));
}

// ---------------------------------------------------------------------------
// Weight swizzle: f32 [Ktot x N] row-major -> bf16 fragment tiles.
// Tile (nt,kt) occupies 512 elems at (nt*KT+kt)*512; within a tile, lane l
// holds column n = nt*16 + (l&15) and K values kt*32 + (l>>4)*8 + (j&7) + (j>>3)*16
// (B-matrix layout: lanes 0-15 hold K={0..7,16..23}, lanes 16-31 hold K={8..15,24..31}).
// ---------------------------------------------------------------------------
__global__ void swizzle_w(const float* __restrict__ src,
                          uint16_t* __restrict__ dst, int Ktot, int N) {
  int id = blockIdx.x * blockDim.x + threadIdx.x;
  int total = Ktot * N;
  if (id >= total) return;
  int tile = id >> 9;
  int lane = (id >> 4) & 31;
  int j    = id & 15;
  int KT   = Ktot >> 5;
  int nt   = tile / KT;
  int kt   = tile - nt * KT;
  int k    = kt * 32 + (lane >> 4) * 8 + (j & 7) + (j >> 3) * 16;
  int n    = nt * 16 + (lane & 15);
  dst[id] = f2b(src[(size_t)k * N + n]);
}

// ---------------------------------------------------------------------------
// GRU recurrence: 32 WGs x 512 threads. WG b owns batch rows [16b,16b+16).
// Wave w owns columns [64w, 64w+64) of z/r/cand/h; h kept in f32 registers.
// ---------------------------------------------------------------------------
__global__ __launch_bounds__(512, 1)
void gru_kernel(const float* __restrict__ x,
                const uint16_t* __restrict__ Wz, const uint16_t* __restrict__ Wr,
                const uint16_t* __restrict__ Wh,
                const float* __restrict__ bz, const float* __restrict__ br,
                const float* __restrict__ bh,
                float* __restrict__ hbuf) {
  __shared__ __attribute__((aligned(16))) uint16_t comb[BT * CSTR];

  const int tid  = threadIdx.x;
  const int lane = tid & 31;
  const int wave = tid >> 5;        // 0..15
  const int half = lane >> 4;       // 0/1
  const int ln   = lane & 15;
  const int rowbase = blockIdx.x * BT;

  // Hoisted per-lane fragment base pointers: tile i lives at +i*TILE_G elems
  // (constant byte offset -> instruction IOFFSET), kt at +(kt<<9) elems.
  const int laneoff = ((wave * 4 * KT_G) << 9) + lane * 16;
  const uint16_t* __restrict__ pzB = Wz + laneoff;
  const uint16_t* __restrict__ prB = Wr + laneoff;
  const uint16_t* __restrict__ phB = Wh + laneoff;
  const uint16_t* arow = &comb[ln * CSTR];
  const int akoff = half * 8;

  int col[4];
  float bzr[4], brr[4], bhr[4];
#pragma unroll
  for (int i = 0; i < 4; ++i) {
    col[i] = wave * 64 + i * 16 + ln;
    bzr[i] = bz[col[i]];
    brr[i] = br[col[i]];
    bhr[i] = bh[col[i]];
  }

  // init LDS: h section = 0, x section = x[:, t=0, :]
  {
    int r  = tid >> 5;
    int c0 = (tid & 31) * 32;
    uint16_t* dsth = &comb[r * CSTR + Iq];
#pragma unroll
    for (int j = 0; j < 32; ++j) dsth[c0 + j] = 0;
    int xc0 = (tid & 31) * 8;
    const float* xs = x + (size_t)(rowbase + r) * (Tq * (size_t)Iq) + xc0;
    uint16_t* dstx = &comb[r * CSTR + xc0];
#pragma unroll
    for (int j = 0; j < 8; ++j) dstx[j] = f2b(xs[j]);
  }

  float hreg[4][8];
#pragma unroll
  for (int i = 0; i < 4; ++i)
#pragma unroll
    for (int v = 0; v < 8; ++v) hreg[i][v] = 0.0f;

  __syncthreads();

  for (int t = 0; t < Tq; ++t) {
    // ---- phase 1: z and r gates, comb = [x_t | h] ---------------------
    v8f zac[4], rac[4];
#pragma unroll
    for (int i = 0; i < 4; ++i) { zac[i] = (v8f)0.0f; rac[i] = (v8f)0.0f; }

#pragma unroll 2
    for (int kt = 0; kt < KT_G; ++kt) {
      v16bf a = load_afrag(arow, kt * 32 + akoff);
      const uint16_t* pz = pzB + (kt << 9);
      const uint16_t* pr = prB + (kt << 9);
#pragma unroll
      for (int i = 0; i < 4; ++i) {
        v16bf b0 = load_bfrag(pz + i * TILE_G);
        zac[i] = wmma_bf16(a, b0, zac[i]);
        v16bf b1 = load_bfrag(pr + i * TILE_G);
        rac[i] = wmma_bf16(a, b1, rac[i]);
      }
    }

    float zv[4][8], rv[4][8];
#pragma unroll
    for (int i = 0; i < 4; ++i)
#pragma unroll
      for (int v = 0; v < 8; ++v) {
        zv[i][v] = sigmoidf_(zac[i][v] + bzr[i]);
        rv[i][v] = sigmoidf_(rac[i][v] + brr[i]);
      }

    __syncthreads();   // all phase-1 reads of comb done

    // write r*h into h section of comb
#pragma unroll
    for (int i = 0; i < 4; ++i)
#pragma unroll
      for (int v = 0; v < 8; ++v) {
        int m = v + half * 8;      // C-layout row for this lane/VGPR
        comb[m * CSTR + Iq + col[i]] = f2b(rv[i][v] * hreg[i][v]);
      }

    __syncthreads();   // r*h visible

    // ---- phase 2: candidate, comb = [x_t | r*h] -----------------------
    v8f cac[4];
#pragma unroll
    for (int i = 0; i < 4; ++i) cac[i] = (v8f)0.0f;

#pragma unroll 2
    for (int kt = 0; kt < KT_G; ++kt) {
      v16bf a = load_afrag(arow, kt * 32 + akoff);
      const uint16_t* ph = phB + (kt << 9);
#pragma unroll
      for (int i = 0; i < 4; ++i) {
        v16bf b = load_bfrag(ph + i * TILE_G);
        cac[i] = wmma_bf16(a, b, cac[i]);
      }
    }

    __syncthreads();   // phase-2 reads done; safe to overwrite comb

    // ---- phase 3: h update (f32), refresh comb ------------------------
#pragma unroll
    for (int i = 0; i < 4; ++i)
#pragma unroll
      for (int v = 0; v < 8; ++v) {
        float cand = tanhf(cac[i][v] + bhr[i]);
        float hn   = hreg[i][v] + zv[i][v] * (cand - hreg[i][v]);
        hreg[i][v] = hn;
        int m = v + half * 8;
        comb[m * CSTR + Iq + col[i]] = f2b(hn);
      }

    if (t + 1 < Tq) {
      int r   = tid >> 5;
      int xc0 = (tid & 31) * 8;
      const float* xs = x + (size_t)(rowbase + r) * (Tq * (size_t)Iq)
                          + (size_t)(t + 1) * Iq + xc0;
      uint16_t* dstx = &comb[r * CSTR + xc0];
#pragma unroll
      for (int j = 0; j < 8; ++j) dstx[j] = f2b(xs[j]);
    }

    __syncthreads();   // comb = [x_{t+1} | h_{t+1}] ready
  }

  // write final hidden state (f32) for the head kernel
#pragma unroll
  for (int i = 0; i < 4; ++i)
#pragma unroll
    for (int v = 0; v < 8; ++v) {
      int m = v + half * 8;
      hbuf[(size_t)(rowbase + m) * Hq + col[i]] = hreg[i][v];
    }
}

// ---------------------------------------------------------------------------
// Head: out = relu(hT @ W1 + b1) @ W2 + b2
// ---------------------------------------------------------------------------
__global__ __launch_bounds__(512, 1)
void head_kernel(const float* __restrict__ hbuf,
                 const uint16_t* __restrict__ W1, const uint16_t* __restrict__ W2,
                 const float* __restrict__ b1, const float* __restrict__ b2,
                 float* __restrict__ out) {
  __shared__ __attribute__((aligned(16))) uint16_t hl[BT * HSTR];

  const int tid  = threadIdx.x;
  const int lane = tid & 31;
  const int wave = tid >> 5;
  const int half = lane >> 4;
  const int ln   = lane & 15;
  const int rowbase = blockIdx.x * BT;

  {
    int r  = tid >> 5;
    int c0 = (tid & 31) * 32;
    const float* src = hbuf + (size_t)(rowbase + r) * Hq + c0;
    uint16_t* dst = &hl[r * HSTR + c0];
#pragma unroll
    for (int j = 0; j < 32; ++j) dst[j] = f2b(src[j]);
  }
  __syncthreads();

  const uint16_t* arow = &hl[ln * HSTR];
  const int akoff = half * 8;

  // GEMM1: [16x1024] @ [1024x1024]; wave w -> tiles 4w..4w+3
  const uint16_t* __restrict__ p1B = W1 + ((wave * 4 * KT_H) << 9) + lane * 16;
  v8f ac[4];
#pragma unroll
  for (int i = 0; i < 4; ++i) ac[i] = (v8f)0.0f;

#pragma unroll 2
  for (int kt = 0; kt < KT_H; ++kt) {
    v16bf a = load_afrag(arow, kt * 32 + akoff);
    const uint16_t* p1 = p1B + (kt << 9);
#pragma unroll
    for (int i = 0; i < 4; ++i) {
      v16bf b = load_bfrag(p1 + i * TILE_H);
      ac[i] = wmma_bf16(a, b, ac[i]);
    }
  }

  int col[4];
  float mv[4][8];
#pragma unroll
  for (int i = 0; i < 4; ++i) {
    col[i] = wave * 64 + i * 16 + ln;
    float bb = b1[col[i]];
#pragma unroll
    for (int v = 0; v < 8; ++v) mv[i][v] = fmaxf(ac[i][v] + bb, 0.0f);
  }

  __syncthreads();   // all GEMM1 reads done; reuse hl for the mid activation
#pragma unroll
  for (int i = 0; i < 4; ++i)
#pragma unroll
    for (int v = 0; v < 8; ++v) {
      int m = v + half * 8;
      hl[m * HSTR + col[i]] = f2b(mv[i][v]);
    }
  __syncthreads();

  // GEMM2: [16x1024] @ [1024x64]; waves 0..3 take one 16-col tile each
  if (wave < 4) {
    const uint16_t* __restrict__ p2B = W2 + ((wave * KT_H) << 9) + lane * 16;
    v8f oc = (v8f)0.0f;
#pragma unroll 2
    for (int kt = 0; kt < KT_H; ++kt) {
      v16bf a = load_afrag(arow, kt * 32 + akoff);
      v16bf b = load_bfrag(p2B + (kt << 9));
      oc = wmma_bf16(a, b, oc);
    }
    int c = wave * 16 + ln;
    float bb = b2[c];
#pragma unroll
    for (int v = 0; v < 8; ++v) {
      int m = v + half * 8;
      out[(size_t)(rowbase + m) * Oq + c] = oc[v] + bb;
    }
  }
}

// ---------------------------------------------------------------------------
extern "C" void kernel_launch(void* const* d_in, const int* in_sizes, int n_in,
                              void* d_out, int out_size, void* d_ws, size_t ws_size,
                              hipStream_t stream) {
  (void)in_sizes; (void)n_in; (void)out_size; (void)ws_size;

  const float* x  = (const float*)d_in[0];
  // d_in[1] = is_missing_mask (unused by reference forward)
  const float* Wz = (const float*)d_in[2];
  const float* bz = (const float*)d_in[3];
  const float* Wr = (const float*)d_in[4];
  const float* br = (const float*)d_in[5];
  const float* Wh = (const float*)d_in[6];
  const float* bh = (const float*)d_in[7];
  const float* W1 = (const float*)d_in[8];
  const float* b1 = (const float*)d_in[9];
  const float* W2 = (const float*)d_in[10];
  const float* b2 = (const float*)d_in[11];
  float* out = (float*)d_out;

  uint8_t* ws = (uint8_t*)d_ws;
  uint16_t* wz_s = (uint16_t*)(ws + WZ_OFF);
  uint16_t* wr_s = (uint16_t*)(ws + WR_OFF);
  uint16_t* wh_s = (uint16_t*)(ws + WH_OFF);
  uint16_t* w1_s = (uint16_t*)(ws + W1_OFF);
  uint16_t* w2_s = (uint16_t*)(ws + W2_OFF);
  float*    hbuf = (float*)(ws + H_OFF);

  const int thr = 256;
  swizzle_w<<<(1280 * 1024 + thr - 1) / thr, thr, 0, stream>>>(Wz, wz_s, 1280, 1024);
  swizzle_w<<<(1280 * 1024 + thr - 1) / thr, thr, 0, stream>>>(Wr, wr_s, 1280, 1024);
  swizzle_w<<<(1280 * 1024 + thr - 1) / thr, thr, 0, stream>>>(Wh, wh_s, 1280, 1024);
  swizzle_w<<<(1024 * 1024 + thr - 1) / thr, thr, 0, stream>>>(W1, w1_s, 1024, 1024);
  swizzle_w<<<(1024 * 64   + thr - 1) / thr, thr, 0, stream>>>(W2, w2_s, 1024, 64);

  gru_kernel<<<Bq / BT, 512, 0, stream>>>(x, wz_s, wr_s, wh_s, bz, br, bh, hbuf);
  head_kernel<<<Bq / BT, 512, 0, stream>>>(hbuf, w1_s, w2_s, b1, b2, out);
}
